// CNN_GRU_GNN_classifier_2_75076028334448
// MI455X (gfx1250) — compile-verified, matrix operand order checked by reading, not run on hardware
//
#include <hip/hip_runtime.h>
#include <hip/hip_bf16.h>
#include <math.h>

typedef __attribute__((ext_vector_type(16))) _Float16 v16h;
typedef __attribute__((ext_vector_type(8)))  float    v8f;
typedef __attribute__((ext_vector_type(4)))  float    f4;
typedef __attribute__((ext_vector_type(4)))  _Float16 h4;
typedef __attribute__((ext_vector_type(8)))  _Float16 h8;

#define BG      8
#define TS      25
#define HIDN    256
#define NPG     2000
#define NNODES  16000
#define E0      256000
#define ETOT    272000
#define BT      200
#define EPSBN   1e-5f

// ---------------------------------------------------------------------------
// Direct 3x3x3 conv (pad 1) + BatchNorm + ReLU.  Small tensors -> VALU.
// ---------------------------------------------------------------------------
__global__ void conv3d_bn_relu(const float* __restrict__ in,
                               const float* __restrict__ w,
                               const float* __restrict__ bias,
                               const float* __restrict__ bg, const float* __restrict__ bb,
                               const float* __restrict__ bm, const float* __restrict__ bv,
                               float* __restrict__ out,
                               int NB, int Cin, int Cout, int D, int H, int W) {
  long total = (long)NB * Cout * D * H * W;
  long idx = (long)blockIdx.x * blockDim.x + threadIdx.x;
  if (idx >= total) return;
  int x = (int)(idx % W); long t = idx / W;
  int y = (int)(t % H); t /= H;
  int z = (int)(t % D); t /= D;
  int co = (int)(t % Cout); int n = (int)(t / Cout);
  float acc = 0.f;
  for (int ci = 0; ci < Cin; ++ci) {
    const float* ip = in + (((long)n * Cin + ci) * D) * H * W;
    const float* wp = w + (((long)co * Cin + ci) * 27);
    for (int kd = 0; kd < 3; ++kd) {
      int zz = z + kd - 1; if (zz < 0 || zz >= D) continue;
      for (int kh = 0; kh < 3; ++kh) {
        int yy = y + kh - 1; if (yy < 0 || yy >= H) continue;
        for (int kw = 0; kw < 3; ++kw) {
          int xx = x + kw - 1; if (xx < 0 || xx >= W) continue;
          acc += ip[((long)zz * H + yy) * W + xx] * wp[kd * 9 + kh * 3 + kw];
        }
      }
    }
  }
  acc += bias[co];
  float sc = bg[co] * rsqrtf(bv[co] + EPSBN);
  acc = (acc - bm[co]) * sc + bb[co];
  out[idx] = fmaxf(acc, 0.f);
}

// ---------------------------------------------------------------------------
// 2x2x2 max-pool, stride 2, configurable padding.
// ---------------------------------------------------------------------------
__global__ void maxpool3d(const float* __restrict__ in, float* __restrict__ out,
                          int NB, int C, int D, int H, int W,
                          int OD, int OH, int OW, int pd, int ph, int pw) {
  long total = (long)NB * C * OD * OH * OW;
  long idx = (long)blockIdx.x * blockDim.x + threadIdx.x;
  if (idx >= total) return;
  int ox = (int)(idx % OW); long t = idx / OW;
  int oy = (int)(t % OH); t /= OH;
  int oz = (int)(t % OD); t /= OD;
  int c = (int)(t % C); int n = (int)(t / C);
  float m = -3.0e38f;
  for (int kd = 0; kd < 2; ++kd) {
    int z = oz * 2 - pd + kd; if (z < 0 || z >= D) continue;
    for (int kh = 0; kh < 2; ++kh) {
      int y = oy * 2 - ph + kh; if (y < 0 || y >= H) continue;
      for (int kw = 0; kw < 2; ++kw) {
        int x = ox * 2 - pw + kw; if (x < 0 || x >= W) continue;
        float v = in[(((long)n * C + c) * D + z) * H * W + (long)y * W + x];
        m = fmaxf(m, v);
      }
    }
  }
  out[idx] = m;
}

// ---------------------------------------------------------------------------
// WMMA GEMM tile pipeline.
//  Block tile 32(M) x 128(N), K stepped by 32, 256 threads = 8 waves (wave32).
//  Each wave owns a 16x32 output strip: two 16x16 WMMA tiles sharing the A
//  fragment (two accumulators, 2 v_wmma per K-step).
//  LDS:  As[32][32] row-major (64B pitch)  -> A fragment = 2x ds_load_b128
//        Bs[128][32] transposed [n][k]     -> B fragment = 2x ds_load_b128
//  Pipeline: global loads for tile k+1 are issued BEFORE the WMMAs of tile k;
//  the f32->f16 convert + LDS store happens AFTER them, so s_wait_loadcnt
//  lands behind the matrix math (HBM latency overlapped).
// ---------------------------------------------------------------------------
struct TileRegs { f4 a; f4 b[4]; };

__device__ __forceinline__ void load_tile(TileRegs& r,
    const float* __restrict__ A, const float* __restrict__ B,
    int M, int N, int K, int btrans, int m0, int n0, int k0, int tid) {
  const int arow = tid >> 3;           // 0..31
  const int akc  = (tid & 7) << 2;     // 0,4,...,28
  const int bn   = tid >> 1;           // 0..127
  const int bkc  = (tid & 1) << 4;     // 0 or 16
  const bool fastA = (m0 + 32 <= M) & (k0 + 32 <= K);
  const bool fastB = (n0 + 128 <= N) & (k0 + 32 <= K);
  if (fastA) {
    const float* ap = A + (long)(m0 + arow) * K + k0 + akc;
    r.a = *(const f4*)ap;                                   // global_load_b128
    if (k0 + 32 < K) __builtin_prefetch(ap + 32, 0, 1);     // global_prefetch_b8
  } else {
#pragma unroll
    for (int j = 0; j < 4; ++j) {
      int gr = m0 + arow, gk = k0 + akc + j;
      r.a[j] = (gr < M && gk < K) ? A[(long)gr * K + gk] : 0.f;
    }
  }
  const int gn = n0 + bn;
  if (fastB) {
    if (btrans) {                                           // contiguous in K
      const float* bp = B + (long)gn * K + k0 + bkc;
#pragma unroll
      for (int q = 0; q < 4; ++q) r.b[q] = *(const f4*)(bp + 4 * q);
    } else {                                                // strided gather
#pragma unroll
      for (int q = 0; q < 4; ++q)
#pragma unroll
        for (int j = 0; j < 4; ++j)
          r.b[q][j] = B[(long)(k0 + bkc + 4 * q + j) * N + gn];
    }
  } else {
#pragma unroll
    for (int q = 0; q < 4; ++q)
#pragma unroll
      for (int j = 0; j < 4; ++j) {
        int gk = k0 + bkc + 4 * q + j;
        float v = 0.f;
        if (gn < N && gk < K)
          v = btrans ? B[(long)gn * K + gk] : B[(long)gk * N + gn];
        r.b[q][j] = v;
      }
  }
}

__device__ __forceinline__ void store_tile(const TileRegs& r,
    _Float16 (*__restrict__ As)[32], _Float16 (*__restrict__ Bs)[32], int tid) {
  const int arow = tid >> 3;
  const int akc  = (tid & 7) << 2;
  const int bn   = tid >> 1;
  const int bkc  = (tid & 1) << 4;
  h4 ah;
  ah[0] = (_Float16)r.a[0]; ah[1] = (_Float16)r.a[1];
  ah[2] = (_Float16)r.a[2]; ah[3] = (_Float16)r.a[3];
  *(h4*)&As[arow][akc] = ah;                                // ds_store_b64
  h8 bh0, bh1;
#pragma unroll
  for (int j = 0; j < 4; ++j) {
    bh0[j]     = (_Float16)r.b[0][j];
    bh0[4 + j] = (_Float16)r.b[1][j];
    bh1[j]     = (_Float16)r.b[2][j];
    bh1[4 + j] = (_Float16)r.b[3][j];
  }
  *(h8*)&Bs[bn][bkc]     = bh0;                             // ds_store_b128
  *(h8*)&Bs[bn][bkc + 8] = bh1;                             // ds_store_b128
}

__global__ __launch_bounds__(256)
void gemm_wmma(const float* __restrict__ A, const float* __restrict__ B,
               float* __restrict__ C, int M, int N, int K, int btrans,
               const float* __restrict__ bias,
               const float* __restrict__ bg, const float* __restrict__ bb,
               const float* __restrict__ bm, const float* __restrict__ bv,
               int relu) {
  __shared__ _Float16 As[2][32][32];     //  4 KB
  __shared__ _Float16 Bs[2][128][32];    // 16 KB
  const int tid  = threadIdx.x;
  const int lane = tid & 31;
  const int wave = tid >> 5;
  const int wr = wave >> 2;              // 0..1 (16-row strip)
  const int wc = wave & 3;               // 0..3 (32-col strip)
  const int m0 = blockIdx.y * 32;
  const int n0 = blockIdx.x * 128;

  // Fragment addressing (ISA 7.12.2 16-bit 16x32 layout):
  //   lanes [0,16):  K chunks 0 (0..7)  and 2 (16..23)
  //   lanes [16,32): K chunks 1 (8..15) and 3 (24..31)
  const int mrow = (wr << 4) + (lane & 15);
  const int ncb  = (wc << 5) + (lane & 15);   // first 16-col tile
  const int kc8  = lane >> 4;                 // 0 or 1

  v8f acc0 = {}, acc1 = {};
  const int nk = (K + 31) >> 5;

  TileRegs t0;
  load_tile(t0, A, B, M, N, K, btrans, m0, n0, 0, tid);
  store_tile(t0, As[0], Bs[0], tid);
  __syncthreads();

  for (int kt = 0; kt < nk; ++kt) {
    const int cur = kt & 1;
    const bool more = (kt + 1 < nk);
    TileRegs nxt;
    if (more)  // issue global loads BEFORE the WMMAs of this step
      load_tile(nxt, A, B, M, N, K, btrans, m0, n0, (kt + 1) << 5, tid);

    union { v16h v; h8 h[2]; } afu, bfu0, bfu1;
    const h8* Ar = (const h8*)&As[cur][mrow][0];
    afu.h[0] = Ar[kc8];                       // ds_load_b128
    afu.h[1] = Ar[2 + kc8];                   // ds_load_b128
    const h8* Br0 = (const h8*)&Bs[cur][ncb][0];
    bfu0.h[0] = Br0[kc8];
    bfu0.h[1] = Br0[2 + kc8];
    const h8* Br1 = (const h8*)&Bs[cur][ncb + 16][0];
    bfu1.h[0] = Br1[kc8];
    bfu1.h[1] = Br1[2 + kc8];

    acc0 = __builtin_amdgcn_wmma_f32_16x16x32_f16(false, afu.v, false, bfu0.v,
                                                  (short)0, acc0, false, false);
    acc1 = __builtin_amdgcn_wmma_f32_16x16x32_f16(false, afu.v, false, bfu1.v,
                                                  (short)0, acc1, false, false);

    if (more)  // convert + LDS-store AFTER the WMMAs (wait lands here)
      store_tile(nxt, As[cur ^ 1], Bs[cur ^ 1], tid);
    __syncthreads();
  }

  // Epilogue: C layout VGPR r -> M = r (lanes 0-15) / r+8 (lanes 16-31)
#pragma unroll
  for (int half = 0; half < 2; ++half) {
    const int col = n0 + (wc << 5) + (half << 4) + (lane & 15);
    if (col >= N) continue;
    float sc = 1.f, sh = 0.f;
    if (bg) { sc = bg[col] * rsqrtf(bv[col] + EPSBN); sh = bb[col] - bm[col] * sc; }
    const float bsv = bias ? bias[col] : 0.f;
#pragma unroll
    for (int r = 0; r < 8; ++r) {
      int row = m0 + (wr << 4) + ((lane < 16) ? r : r + 8);
      if (row < M) {
        float v = (half ? acc1[r] : acc0[r]) + bsv;
        if (bg)   v = v * sc + sh;
        if (relu) v = fmaxf(v, 0.f);
        C[(long)row * N + col] = v;
      }
    }
  }
}

// ---------------------------------------------------------------------------
// GRU recurrence: gx (B,T,3H) precomputed with WMMA; one workgroup per batch
// element scans T sequentially keeping h in LDS.
// ---------------------------------------------------------------------------
__global__ __launch_bounds__(HIDN)
void gru_scan(const float* __restrict__ gx, const float* __restrict__ whh,
              const float* __restrict__ bhh, float* __restrict__ ys) {
  __shared__ float h[HIDN];
  const int b = blockIdx.x;
  const int n = threadIdx.x;
  h[n] = 0.f;
  __syncthreads();
  for (int t = 0; t < TS; ++t) {
    float gr = bhh[n], gz = bhh[HIDN + n], gn = bhh[2 * HIDN + n];
    const float* wr = whh + (long)n * HIDN;
    const float* wz = whh + (long)(HIDN + n) * HIDN;
    const float* wn = whh + (long)(2 * HIDN + n) * HIDN;
    for (int k = 0; k < HIDN; ++k) {
      float hv = h[k];
      gr += wr[k] * hv; gz += wz[k] * hv; gn += wn[k] * hv;
    }
    const float* gxt = gx + ((long)b * TS + t) * 3 * HIDN;
    float r = 1.f / (1.f + __expf(-(gxt[n] + gr)));
    float z = 1.f / (1.f + __expf(-(gxt[HIDN + n] + gz)));
    float nn = tanhf(gxt[2 * HIDN + n] + r * gn);
    float hnew = (1.f - z) * nn + z * h[n];
    __syncthreads();
    h[n] = hnew;
    ys[((long)b * TS + t) * HIDN + n] = hnew;
    __syncthreads();
  }
}

// ---------------------------------------------------------------------------
// GAT layer 0 factorized transform: h0 = node_x @ W[:3] + (enc @ W[3:])[graph]
// ---------------------------------------------------------------------------
__global__ void gat0_combine(const float* __restrict__ nodex,
                             const float* __restrict__ w0,
                             const float* __restrict__ encpart,
                             float* __restrict__ h) {
  const int node = blockIdx.x;
  const int f = threadIdx.x;
  const int g = node / NPG;
  float v = encpart[(long)g * HIDN + f];
  v += nodex[node * 3 + 0] * w0[f];
  v += nodex[node * 3 + 1] * w0[HIDN + f];
  v += nodex[node * 3 + 2] * w0[2 * HIDN + f];
  h[(long)node * HIDN + f] = v;
}

__global__ void gat_att(const float* __restrict__ h,
                        const float* __restrict__ av_s, const float* __restrict__ av_d,
                        float* __restrict__ a_s, float* __restrict__ a_d, int F) {
  int n = blockIdx.x * blockDim.x + threadIdx.x;
  if (n >= NNODES) return;
  float s = 0.f, d = 0.f;
  for (int f = 0; f < F; ++f) {
    float hv = h[(long)n * F + f];
    s += hv * av_s[f]; d += hv * av_d[f];
  }
  a_s[n] = s; a_d[n] = d;
}

__global__ void fill_f32(float* __restrict__ p, long n, float v) {
  long i = (long)blockIdx.x * blockDim.x + threadIdx.x;
  if (i < n) p[i] = v;
}

__device__ __forceinline__ void atomicMaxF(float* addr, float val) {
  unsigned int* ai = (unsigned int*)addr;
  unsigned int old = *ai, assumed;
  do {
    assumed = old;
    if (__uint_as_float(assumed) >= val) break;
    old = atomicCAS(ai, assumed, __float_as_uint(val));
  } while (old != assumed);
}

__global__ void edge_max(const int* __restrict__ ei,
                         const float* __restrict__ a_s, const float* __restrict__ a_d,
                         float* __restrict__ e, float* __restrict__ mx) {
  int i = blockIdx.x * blockDim.x + threadIdx.x;
  if (i >= ETOT) return;
  int s = (i < E0) ? ei[i]      : (i - E0);
  int d = (i < E0) ? ei[E0 + i] : (i - E0);
  float v = a_s[s] + a_d[d];
  v = (v > 0.f) ? v : 0.2f * v;                // leaky_relu(0.2)
  e[i] = v;
  atomicMaxF(&mx[d], v);
}

__global__ void edge_exp(const int* __restrict__ ei,
                         const float* __restrict__ e, const float* __restrict__ mx,
                         float* __restrict__ exb, float* __restrict__ ssum,
                         float* __restrict__ deg) {
  int i = blockIdx.x * blockDim.x + threadIdx.x;
  if (i >= ETOT) return;
  int d = (i < E0) ? ei[E0 + i] : (i - E0);
  float ex = __expf(e[i] - mx[d]);
  exb[i] = ex;
  atomicAdd(&ssum[d], ex);
  atomicAdd(&deg[d], 1.f);
}

__global__ void edge_agg(const int* __restrict__ ei,
                         const float* __restrict__ exb, const float* __restrict__ ssum,
                         const float* __restrict__ ht, float* __restrict__ agg, int F) {
  int i = blockIdx.x;
  int s = (i < E0) ? ei[i]      : (i - E0);
  int d = (i < E0) ? ei[E0 + i] : (i - E0);
  float alpha = exb[i] / ssum[d];
  for (int f = threadIdx.x; f < F; f += blockDim.x)
    atomicAdd(&agg[(long)d * F + f], alpha * ht[(long)s * F + f]);
}

__global__ void gat_finalize(const float* __restrict__ agg, const float* __restrict__ deg,
                             const float* __restrict__ bvec, float* __restrict__ outp,
                             int F, int relu) {
  long idx = (long)blockIdx.x * blockDim.x + threadIdx.x;
  if (idx >= (long)NNODES * F) return;
  int node = (int)(idx / F);
  int f = (int)(idx % F);
  float v = agg[idx] / deg[node] + bvec[f];
  if (relu) v = fmaxf(v, 0.f);
  outp[idx] = v;
}

__global__ void softmax2(const float* __restrict__ h, float* __restrict__ out) {
  int n = blockIdx.x * blockDim.x + threadIdx.x;
  if (n >= NNODES) return;
  float a = h[n * 2], b = h[n * 2 + 1];
  float m = fmaxf(a, b);
  float ea = __expf(a - m), eb = __expf(b - m);
  float s = ea + eb;
  out[n * 2] = ea / s;
  out[n * 2 + 1] = eb / s;
}

__global__ void copy_i32(const int* __restrict__ s, int* __restrict__ d, int n) {
  int i = blockIdx.x * blockDim.x + threadIdx.x;
  if (i < n) d[i] = s[i];
}

// ---------------------------------------------------------------------------
extern "C" void kernel_launch(void* const* d_in, const int* in_sizes, int n_in,
                              void* d_out, int out_size, void* d_ws, size_t ws_size,
                              hipStream_t stream) {
  (void)in_sizes; (void)n_in; (void)out_size; (void)ws_size;
#define IN_F(i) ((const float*)d_in[(i)])
#define IN_I(i) ((const int*)d_in[(i)])
  const float* X      = IN_F(0);
  const float* nodex  = IN_F(1);
  const int*   ei     = IN_I(2);
  const int*   ny     = IN_I(3);
  const float* conv1w = IN_F(4);  const float* conv1b = IN_F(5);
  const float* bn1[4] = {IN_F(6), IN_F(7), IN_F(8), IN_F(9)};
  const float* conv2w = IN_F(10); const float* conv2b = IN_F(11);
  const float* bn2[4] = {IN_F(12), IN_F(13), IN_F(14), IN_F(15)};
  const float* conv3w = IN_F(16); const float* conv3b = IN_F(17);
  const float* bn3[4] = {IN_F(18), IN_F(19), IN_F(20), IN_F(21)};
  const float* fc1w = IN_F(22); const float* fc1b = IN_F(23);
  const float* bn4[4] = {IN_F(24), IN_F(25), IN_F(26), IN_F(27)};
  const float* fc2w = IN_F(28); const float* fc2b = IN_F(29);
  const float* bn5[4] = {IN_F(30), IN_F(31), IN_F(32), IN_F(33)};
  const float* g0wih = IN_F(34); const float* g0whh = IN_F(35);
  const float* g0bih = IN_F(36); const float* g0bhh = IN_F(37);
  const float* g1wih = IN_F(38); const float* g1whh = IN_F(39);
  const float* g1bih = IN_F(40); const float* g1bhh = IN_F(41);
  const float* gatw[7]; const float* gatas[7]; const float* gatad[7]; const float* gatb[7];
  for (int i = 0; i < 7; ++i) {
    gatw[i]  = IN_F(42 + 4 * i);
    gatas[i] = IN_F(43 + 4 * i);
    gatad[i] = IN_F(44 + 4 * i);
    gatb[i]  = IN_F(45 + 4 * i);
  }

  // ---- workspace carve-up ----
  size_t off = 0;
  auto alloc = [&](size_t elems) -> float* {
    float* p = (float*)((char*)d_ws + off);
    off += ((elems * 4 + 255) & ~(size_t)255);
    return p;
  };
  float* b_c1   = alloc((size_t)BT * 64 * 5 * 6 * 6);    // conv1 out
  float* b_c2   = alloc((size_t)BT * 64 * 5 * 6 * 6);    // conv2 out
  float* b_p1   = alloc((size_t)BT * 64 * 3 * 4 * 4);    // pool1 out
  float* b_c3   = alloc((size_t)BT * 256 * 3 * 4 * 4);   // conv3 out
  float* b_p2   = alloc((size_t)BT * 2048);              // pool2 flat
  float* b_fc1  = alloc((size_t)BT * 512);
  float* b_fc2  = alloc((size_t)BT * HIDN);
  float* b_gx   = alloc((size_t)BT * 3 * HIDN);
  float* b_gru0 = alloc((size_t)BT * HIDN);
  float* b_gru1 = alloc((size_t)BT * HIDN);              // == enc (8 x 6400)
  float* b_enc  = alloc((size_t)BG * HIDN);              // enc @ W0[3:]
  float* hA     = alloc((size_t)NNODES * HIDN);          // GAT features / output
  float* hB     = alloc((size_t)NNODES * HIDN);          // transformed h
  float* b_agg  = alloc((size_t)NNODES * HIDN);
  float* b_e    = alloc((size_t)ETOT);
  float* b_ex   = alloc((size_t)ETOT);
  float* b_mx   = alloc((size_t)NNODES);
  float* b_ss   = alloc((size_t)NNODES);
  float* b_deg  = alloc((size_t)NNODES);
  float* b_as   = alloc((size_t)NNODES);
  float* b_ad   = alloc((size_t)NNODES);

  const int TPB = 256;
  auto cdiv = [](long a, long b) { return (unsigned)((a + b - 1) / b); };
  auto gemm = [&](const float* A, const float* Bm, float* Cm, int M, int N, int K,
                  int btrans, const float* bias, const float* const* bn, int relu) {
    dim3 grid(cdiv(N, 128), cdiv(M, 32));
    gemm_wmma<<<grid, 256, 0, stream>>>(A, Bm, Cm, M, N, K, btrans, bias,
                                        bn ? bn[0] : nullptr, bn ? bn[1] : nullptr,
                                        bn ? bn[2] : nullptr, bn ? bn[3] : nullptr, relu);
  };

  // ---- CNN trunk ----
  conv3d_bn_relu<<<cdiv((long)BT * 64 * 5 * 6 * 6, TPB), TPB, 0, stream>>>(
      X, conv1w, conv1b, bn1[0], bn1[1], bn1[2], bn1[3], b_c1, BT, 5, 64, 5, 6, 6);
  conv3d_bn_relu<<<cdiv((long)BT * 64 * 5 * 6 * 6, TPB), TPB, 0, stream>>>(
      b_c1, conv2w, conv2b, bn2[0], bn2[1], bn2[2], bn2[3], b_c2, BT, 64, 64, 5, 6, 6);
  maxpool3d<<<cdiv((long)BT * 64 * 3 * 4 * 4, TPB), TPB, 0, stream>>>(
      b_c2, b_p1, BT, 64, 5, 6, 6, 3, 4, 4, 1, 1, 1);
  conv3d_bn_relu<<<cdiv((long)BT * 256 * 3 * 4 * 4, TPB), TPB, 0, stream>>>(
      b_p1, conv3w, conv3b, bn3[0], bn3[1], bn3[2], bn3[3], b_c3, BT, 64, 256, 3, 4, 4);
  maxpool3d<<<cdiv((long)BT * 2048, TPB), TPB, 0, stream>>>(
      b_c3, b_p2, BT, 256, 3, 4, 4, 2, 2, 2, 1, 0, 0);

  // ---- FC (WMMA) ----
  gemm(b_p2, fc1w, b_fc1, BT, 512, 2048, 1, fc1b, bn4, 1);
  gemm(b_fc1, fc2w, b_fc2, BT, HIDN, 512, 1, fc2b, bn5, 1);

  // ---- GRU (input projection via WMMA, recurrence scan in LDS) ----
  gemm(b_fc2, g0wih, b_gx, BT, 3 * HIDN, HIDN, 1, g0bih, nullptr, 0);
  gru_scan<<<BG, HIDN, 0, stream>>>(b_gx, g0whh, g0bhh, b_gru0);
  gemm(b_gru0, g1wih, b_gx, BT, 3 * HIDN, HIDN, 1, g1bih, nullptr, 0);
  gru_scan<<<BG, HIDN, 0, stream>>>(b_gx, g1whh, g1bhh, b_gru1);  // enc = (8, 6400)

  // ---- GAT stack ----
  for (int i = 0; i < 7; ++i) {
    const int F = (i < 6) ? HIDN : 2;
    if (i == 0) {
      // enc-part: (8 x 6400) @ W0[3:,:] (row-major 6400x256), then broadcast-add
      gemm(b_gru1, gatw[0] + 3 * HIDN, b_enc, BG, HIDN, TS * HIDN, 0, nullptr, nullptr, 0);
      gat0_combine<<<NNODES, HIDN, 0, stream>>>(nodex, gatw[0], b_enc, hB);
    } else {
      gemm(hA, gatw[i], hB, NNODES, F, HIDN, 0, nullptr, nullptr, 0);
    }
    gat_att<<<cdiv(NNODES, TPB), TPB, 0, stream>>>(hB, gatas[i], gatad[i], b_as, b_ad, F);
    fill_f32<<<cdiv(NNODES, TPB), TPB, 0, stream>>>(b_mx, NNODES, -3.0e38f);
    fill_f32<<<cdiv(NNODES, TPB), TPB, 0, stream>>>(b_ss, NNODES, 0.f);
    fill_f32<<<cdiv(NNODES, TPB), TPB, 0, stream>>>(b_deg, NNODES, 0.f);
    fill_f32<<<cdiv((long)NNODES * F, TPB), TPB, 0, stream>>>(b_agg, (long)NNODES * F, 0.f);
    edge_max<<<cdiv(ETOT, TPB), TPB, 0, stream>>>(ei, b_as, b_ad, b_e, b_mx);
    edge_exp<<<cdiv(ETOT, TPB), TPB, 0, stream>>>(ei, b_e, b_mx, b_ex, b_ss, b_deg);
    edge_agg<<<ETOT, TPB, 0, stream>>>(ei, b_ex, b_ss, hB, b_agg, F);
    gat_finalize<<<cdiv((long)NNODES * F, TPB), TPB, 0, stream>>>(
        b_agg, b_deg, gatb[i], hA, F, (i < 6) ? 1 : 0);
  }

  // ---- softmax + outputs (y_pred f32, node_y bit-copied) ----
  softmax2<<<cdiv(NNODES, TPB), TPB, 0, stream>>>(hA, (float*)d_out);
  copy_i32<<<cdiv(NNODES, TPB), TPB, 0, stream>>>(ny, (int*)d_out + 2 * NNODES, NNODES);
#undef IN_F
#undef IN_I
}
